// channel_att_block_52999896433173
// MI455X (gfx1250) — matrix operand order repaired and checked by reading, not compiled, not run on hardware
//
#include <hip/hip_runtime.h>
#include <hip/hip_bf16.h>

// ---- problem constants (match reference) ----
constexpr int nB = 32, nC = 128, nS = 512, nD = 512, nDFF = 2048;
constexpr int nH = 8, nL = 3, nZ = 512, nDH = 64;
constexpr float kScale = 0.125f;           // DH^-0.5
constexpr float kBeta  = 0.1f;

typedef __attribute__((ext_vector_type(16))) __bf16 v16bf;
typedef __attribute__((ext_vector_type(8)))  __bf16 v8bf;
typedef __attribute__((ext_vector_type(8)))  float  v8f;

static __device__ __forceinline__ __bf16 f2bf(float f) { return (__bf16)f; }

// A operand: lane holds row (lane&15); K = hi*16+{0..15} contiguous.
// 'a0' already includes +hi*16 floats.
static __device__ __forceinline__ v16bf load_a_f32(const float* __restrict__ a0, int k) {
  float4 x0 = *(const float4*)(a0 + k);
  float4 x1 = *(const float4*)(a0 + k + 4);
  float4 x2 = *(const float4*)(a0 + k + 8);
  float4 x3 = *(const float4*)(a0 + k + 12);
  v16bf v;
  v[0]=f2bf(x0.x); v[1]=f2bf(x0.y); v[2] =f2bf(x0.z); v[3] =f2bf(x0.w);
  v[4]=f2bf(x1.x); v[5]=f2bf(x1.y); v[6] =f2bf(x1.z); v[7] =f2bf(x1.w);
  v[8]=f2bf(x2.x); v[9]=f2bf(x2.y); v[10]=f2bf(x2.z); v[11]=f2bf(x2.w);
  v[12]=f2bf(x3.x);v[13]=f2bf(x3.y);v[14]=f2bf(x3.z); v[15]=f2bf(x3.w);
  return v;
}

// B operand: lane holds column (lane&15); K = hi*16+{0..15} contiguous.
// 'b0' already includes +hi*16 elements.
static __device__ __forceinline__ v16bf load_b_bf16(const __bf16* __restrict__ b0, int k) {
  v8bf lo = *(const v8bf*)(b0 + k);
  v8bf hi = *(const v8bf*)(b0 + k + 8);
  v16bf v;
#pragma unroll
  for (int i = 0; i < 8; ++i) { v[i] = lo[i]; v[8 + i] = hi[i]; }
  return v;
}

static __device__ __forceinline__ v16bf load_b_f32(const float* __restrict__ b0, int k) {
  float4 x0 = *(const float4*)(b0 + k);
  float4 x1 = *(const float4*)(b0 + k + 4);
  float4 x2 = *(const float4*)(b0 + k + 8);
  float4 x3 = *(const float4*)(b0 + k + 12);
  v16bf v;
  v[0]=f2bf(x0.x); v[1]=f2bf(x0.y); v[2] =f2bf(x0.z); v[3] =f2bf(x0.w);
  v[4]=f2bf(x1.x); v[5]=f2bf(x1.y); v[6] =f2bf(x1.z); v[7] =f2bf(x1.w);
  v[8]=f2bf(x2.x); v[9]=f2bf(x2.y); v[10]=f2bf(x2.z); v[11]=f2bf(x2.w);
  v[12]=f2bf(x3.x);v[13]=f2bf(x3.y);v[14]=f2bf(x3.z); v[15]=f2bf(x3.w);
  return v;
}

// One 16-byte async global->LDS copy (GLOBAL_LOAD_ASYNC_TO_LDS_B128, tracked by ASYNCcnt).
static __device__ __forceinline__ void async_copy16(unsigned lds_off, const float* g) {
  asm volatile("global_load_async_to_lds_b128 %0, %1, off"
               :: "v"(lds_off), "v"((unsigned long)g)
               : "memory");
}

// ---------------- generic batched GEMM: C = A(f32) x Wt(bf16,[N,K]) + bias ----------------
// Each wave computes a 32x64 macro-tile = 2x4 WMMA tiles (8 v_wmma per 32-deep K step).
// All 8 waves of a block share the same 32-row A panel (tiles are 8 consecutive column
// blocks), so the fp32 A panel (32x32 = 4KB) is staged into LDS with async copies,
// double-buffered with s_wait_asynccnt, and each wave reads its fragments via ds_load.
// NOTE: launch grids guarantee tiles % 8 == 0 and gridDim.x == tiles/8 (no idle waves).
__global__ void gemm_bf16w(const float* __restrict__ A, long lda, long sA,
                           const __bf16* __restrict__ Wt, long sW,
                           const float* __restrict__ bias, long sBias,
                           float* __restrict__ Cout, long ldc, long sC,
                           int Nb, int K, int tiles)   // Nb = N/64
{
  __shared__ float ldsA[2][32 * 32];                 // two 4KB A panels
  (void)tiles;
  int t = (int)threadIdx.x;
  int wave = t >> 5;
  int tile = blockIdx.x * 8 + wave;
  long batch = blockIdx.y;
  int tm = tile / Nb, tn = tile - tm * Nb;           // tm is block-uniform
  int lane = t & 31, mi = lane & 15, hi = lane >> 4;

  // cooperative A staging: thread t copies 16B chunk (row t/8, cols (t&7)*4..+3)
  const float* gA = A + batch * sA + (long)(tm * 32 + (t >> 3)) * lda + (t & 7) * 4;
  unsigned ldsBase = (unsigned)(unsigned long)(const void*)&ldsA[0][0];
  unsigned myLds   = ldsBase + (unsigned)t * 16u;    // row*128 + chunk*16 == t*16

  const __bf16* b0 = Wt + batch * sW + (long)(tn * 64 + mi) * K + hi * 16;

  async_copy16(myLds, gA);                           // prologue: buffer 0 <- k=0

  v8f acc[2][4] = {};
  for (int k = 0; k < K; k += 32) {
    int cur = (k >> 5) & 1;
    bool more = (k + 32) < K;
    if (more) async_copy16(ldsBase + (unsigned)((cur ^ 1) * 4096) + (unsigned)t * 16u,
                           gA + k + 32);
    __builtin_prefetch(b0 + k + 32, 0, 0);           // global_prefetch_b8 for B stream
    if (more) asm volatile("s_wait_asynccnt 0x1" ::: "memory");
    else      asm volatile("s_wait_asynccnt 0x0" ::: "memory");
    __syncthreads();                                 // panel 'cur' visible to all waves

    const float* ap = &ldsA[cur][mi * 32 + hi * 16];
    v16bf av0 = load_a_f32(ap, 0);                   // ds_load_b128 x4
    v16bf av1 = load_a_f32(ap + 16 * 32, 0);
    v16bf bv[4];
#pragma unroll
    for (int j = 0; j < 4; ++j) bv[j] = load_b_bf16(b0 + (long)j * 16 * K, k);
#pragma unroll
    for (int j = 0; j < 4; ++j) {
      acc[0][j] = __builtin_amdgcn_wmma_f32_16x16x32_bf16(false, av0, false, bv[j],
                                                          (short)0, acc[0][j], false, false);
      acc[1][j] = __builtin_amdgcn_wmma_f32_16x16x32_bf16(false, av1, false, bv[j],
                                                          (short)0, acc[1][j], false, false);
    }
    __syncthreads();                                 // done reading 'cur' before refill
  }
#pragma unroll
  for (int i = 0; i < 2; ++i) {
#pragma unroll
    for (int j = 0; j < 4; ++j) {
      int col = tn * 64 + j * 16 + mi;
      float bc = bias ? bias[batch * sBias + col] : 0.0f;
      float* cp = Cout + batch * sC + (long)(tm * 32 + i * 16 + hi * 8) * ldc + col;
#pragma unroll
      for (int r = 0; r < 8; ++r) cp[(long)r * ldc] = acc[i][j][r] + bc;
    }
  }
}

// ---------------- attention GEMM, batched over (b,h) via grid.y/z ----------------
// Same 32x64 (2x4-tile) register blocking; direct global loads (no LDS, waves may idle).
// BCOL=true : B supplied column-contiguous (scores = Q*K^T, columns are K-rows)
// BCOL=false: B row-major with stride ldb (O = W*V)
template <bool BCOL>
__global__ void attn_gemm(const float* __restrict__ A, long lda, long sAb, long sAh,
                          const float* __restrict__ Bm, long ldb, long sBb, long sBh,
                          float* __restrict__ Cout, long ldc, long sCb, long sCh,
                          int Nb, int K, float alpha, int tiles)
{
  int wave = (int)(threadIdx.x >> 5);
  int tile = blockIdx.x * (blockDim.x >> 5) + wave;
  if (tile >= tiles) return;                  // wave-uniform
  long b = blockIdx.y, h = blockIdx.z;
  int tm = tile / Nb, tn = tile - tm * Nb;
  int lane = threadIdx.x & 31, mi = lane & 15, hi = lane >> 4;

  const float* a0 = A + b * sAb + h * sAh + (long)(tm * 32 + mi) * lda + hi * 16;
  v8f acc[2][4] = {};
  if (BCOL) {
    const float* b0 = Bm + b * sBb + h * sBh + (long)(tn * 64 + mi) * ldb + hi * 16;
    for (int k = 0; k < K; k += 32) {
      __builtin_prefetch(a0 + k + 32, 0, 0);
      __builtin_prefetch(b0 + k + 32, 0, 0);
      v16bf av0 = load_a_f32(a0, k);
      v16bf av1 = load_a_f32(a0 + 16 * lda, k);
      v16bf bv[4];
#pragma unroll
      for (int j = 0; j < 4; ++j) bv[j] = load_b_f32(b0 + (long)j * 16 * ldb, k);
#pragma unroll
      for (int j = 0; j < 4; ++j) {
        acc[0][j] = __builtin_amdgcn_wmma_f32_16x16x32_bf16(false, av0, false, bv[j],
                                                            (short)0, acc[0][j], false, false);
        acc[1][j] = __builtin_amdgcn_wmma_f32_16x16x32_bf16(false, av1, false, bv[j],
                                                            (short)0, acc[1][j], false, false);
      }
    }
  } else {
    const float* b0 = Bm + b * sBb + h * sBh + (tn * 64 + mi);
    for (int k = 0; k < K; k += 32) {
      v16bf av0 = load_a_f32(a0, k);
      v16bf av1 = load_a_f32(a0 + 16 * lda, k);
      int kb = k + hi * 16;
      v16bf bv[4];
#pragma unroll
      for (int j = 0; j < 4; ++j)
#pragma unroll
        for (int e = 0; e < 16; ++e) bv[j][e] = f2bf(b0[(long)(kb + e) * ldb + j * 16]);
#pragma unroll
      for (int j = 0; j < 4; ++j) {
        acc[0][j] = __builtin_amdgcn_wmma_f32_16x16x32_bf16(false, av0, false, bv[j],
                                                            (short)0, acc[0][j], false, false);
        acc[1][j] = __builtin_amdgcn_wmma_f32_16x16x32_bf16(false, av1, false, bv[j],
                                                            (short)0, acc[1][j], false, false);
      }
    }
  }
#pragma unroll
  for (int i = 0; i < 2; ++i) {
#pragma unroll
    for (int j = 0; j < 4; ++j) {
      int col = tn * 64 + j * 16 + mi;
      float* cp = Cout + b * sCb + h * sCh + (long)(tm * 32 + i * 16 + hi * 8) * ldc + col;
#pragma unroll
      for (int r = 0; r < 8; ++r) cp[(long)r * ldc] = acc[i][j][r] * alpha;
    }
  }
}

// ---------------- weight pack: fp32 [K,N] (row-major) -> bf16 [N,K] ----------------
__global__ void pack_wT(const float* __restrict__ W, __bf16* __restrict__ Wt, int K, int N)
{
  long batch = blockIdx.y;
  const float* w = W + batch * (long)K * N;
  __bf16* wt = Wt + batch * (long)K * N;
  long total = (long)K * N;
  for (long i = blockIdx.x * (long)blockDim.x + threadIdx.x; i < total;
       i += (long)gridDim.x * blockDim.x) {
    long n = i / K, k = i - n * K;
    wt[i] = f2bf(w[k * (long)N + n]);
  }
}

// ---------------- softmax over rows of length 128, in place ----------------
__global__ void softmax128(float* __restrict__ s)
{
  __shared__ float red[128];
  int t = threadIdx.x;
  float* p = s + (long)blockIdx.x * 128;
  float v = p[t];
  red[t] = v; __syncthreads();
  for (int o = 64; o > 0; o >>= 1) { if (t < o) red[t] = fmaxf(red[t], red[t + o]); __syncthreads(); }
  float m = red[0]; __syncthreads();
  float e = expf(v - m);
  red[t] = e; __syncthreads();
  for (int o = 64; o > 0; o >>= 1) { if (t < o) red[t] += red[t + o]; __syncthreads(); }
  p[t] = e / red[0];
}

// ---------------- y = o + qe; BatchNorm1d over (B, D) per channel ----------------
__global__ void add_bn(const float* __restrict__ o, const float* __restrict__ qe,
                       float* __restrict__ y, const float* __restrict__ g,
                       const float* __restrict__ bt)
{
  __shared__ float r1[256], r2[256];
  int c = blockIdx.x, t = threadIdx.x;
  float s = 0.f, s2 = 0.f;
  for (int i = t; i < nB * nD; i += 256) {
    int b = i >> 9, d = i & (nD - 1);
    long a = ((long)b * nC + c) * nD + d;
    float v = o[a] + qe[a];
    s += v; s2 += v * v;
  }
  r1[t] = s; r2[t] = s2; __syncthreads();
  for (int off = 128; off > 0; off >>= 1) {
    if (t < off) { r1[t] += r1[t + off]; r2[t] += r2[t + off]; }
    __syncthreads();
  }
  float mu  = r1[0] * (1.f / (nB * nD));
  float var = r2[0] * (1.f / (nB * nD)) - mu * mu;
  float inv = rsqrtf(var + 1e-5f);
  float gg = g[c], bb = bt[c];
  for (int i = t; i < nB * nD; i += 256) {
    int b = i >> 9, d = i & (nD - 1);
    long a = ((long)b * nC + c) * nD + d;
    float v = o[a] + qe[a];
    y[a] = (v - mu) * inv * gg + bb;
  }
}

// ---------------- GEGLU: ff = a * gelu_exact(g), h=[rows,2048] -> ff=[rows,1024] ----------------
__global__ void geglu(const float* __restrict__ h, float* __restrict__ ff)
{
  long i = blockIdx.x * (long)blockDim.x + threadIdx.x;
  long total = (long)nB * nC * (nDFF / 2);
  if (i >= total) return;
  long row = i / (nDFF / 2), j = i - row * (nDFF / 2);
  float a = h[row * nDFF + j];
  float g = h[row * nDFF + nDFF / 2 + j];
  float gl = 0.5f * g * (1.f + erff(g * 0.70710678118654752f));
  ff[i] = a * gl;
}

// ---------------- x = LayerNorm(y + ff2) over D ----------------
__global__ void add_ln(const float* __restrict__ y, const float* __restrict__ ff2,
                       float* __restrict__ x, const float* __restrict__ g,
                       const float* __restrict__ bt)
{
  __shared__ float r1[256], r2[256];
  long row = blockIdx.x; int t = threadIdx.x;
  float v0 = y[row * nD + t]       + ff2[row * nD + t];
  float v1 = y[row * nD + t + 256] + ff2[row * nD + t + 256];
  r1[t] = v0 + v1; r2[t] = v0 * v0 + v1 * v1; __syncthreads();
  for (int off = 128; off > 0; off >>= 1) {
    if (t < off) { r1[t] += r1[t + off]; r2[t] += r2[t + off]; }
    __syncthreads();
  }
  float mu  = r1[0] * (1.f / nD);
  float var = r2[0] * (1.f / nD) - mu * mu;
  float inv = rsqrtf(var + 1e-5f);
  x[row * nD + t]       = (v0 - mu) * inv * g[t]       + bt[t];
  x[row * nD + t + 256] = (v1 - mu) * inv * g[t + 256] + bt[t + 256];
}

// ---------------- KL: per-(b,c) row sums (stage 1, deterministic) ----------------
// log_q - log_p summed over z:  -eps^2/2 - log(sigma) + z^2/2
__global__ void kl_rows(const float* __restrict__ mean, const float* __restrict__ sraw,
                        const float* __restrict__ eps, float* __restrict__ rows)
{
  __shared__ float r[256];
  long row = blockIdx.x; int t = threadIdx.x;
  float acc = 0.f;
  for (int z = t; z < nZ; z += 256) {
    long a = row * nZ + z;
    float m = mean[a], sr = sraw[a], e = eps[a];
    float sp = sr > 20.f ? sr : log1pf(expf(sr));
    float sc = sp + 1e-8f;
    float zz = m + sc * e;
    acc += -0.5f * e * e - logf(sc) + 0.5f * zz * zz;
  }
  r[t] = acc; __syncthreads();
  for (int off = 128; off > 0; off >>= 1) { if (t < off) r[t] += r[t + off]; __syncthreads(); }
  if (t == 0) rows[row] = r[0];
}

// ---------------- KL stage 2: single-block deterministic reduce, accumulate ----------------
__global__ void kl_reduce(const float* __restrict__ rows, float* __restrict__ kl)
{
  __shared__ float r[256];
  int t = threadIdx.x;
  float s = 0.f;
  for (int i = t; i < nB * nC; i += 256) s += rows[i];
  r[t] = s; __syncthreads();
  for (int off = 128; off > 0; off >>= 1) { if (t < off) r[t] += r[t + off]; __syncthreads(); }
  if (t == 0) *kl += r[0] * (kBeta / ((float)nB * nC * nL));
}

__global__ void set_zero1(float* p) { if (threadIdx.x == 0 && blockIdx.x == 0) *p = 0.f; }

// =====================================================================================
extern "C" void kernel_launch(void* const* d_in, const int* in_sizes, int n_in,
                              void* d_out, int out_size, void* d_ws, size_t ws_size,
                              hipStream_t stream)
{
  (void)in_sizes; (void)n_in; (void)out_size; (void)ws_size;
  const float* q       = (const float*)d_in[0];
  const float* inp     = (const float*)d_in[1];
  const float* embed_w = (const float*)d_in[2];
  const float* embed_b = (const float*)d_in[3];
  const float* Wq      = (const float*)d_in[4];
  const float* bq      = (const float*)d_in[5];
  const float* Wk      = (const float*)d_in[6];
  const float* bk      = (const float*)d_in[7];
  const float* Wv      = (const float*)d_in[8];
  const float* bv      = (const float*)d_in[9];
  const float* Wo      = (const float*)d_in[10];
  const float* bo      = (const float*)d_in[11];
  const float* bn_g    = (const float*)d_in[12];
  const float* bn_b    = (const float*)d_in[13];
  const float* ln_g    = (const float*)d_in[14];
  const float* ln_b    = (const float*)d_in[15];
  const float* f1_w    = (const float*)d_in[16];
  const float* f1_b    = (const float*)d_in[17];
  const float* f2_w    = (const float*)d_in[18];
  const float* f2_b    = (const float*)d_in[19];
  const float* loc_w   = (const float*)d_in[20];
  const float* loc_b   = (const float*)d_in[21];
  const float* sc_w    = (const float*)d_in[22];
  const float* sc_b    = (const float*)d_in[23];
  const float* eps     = (const float*)d_in[24];

  float* xout = (float*)d_out;                       // [B,C,D] final x lives in d_out
  float* klp  = xout + (long)nB * nC * nD;           // scalar kl

  // ---- workspace layout (256B aligned slices) ----
  char* wsb = (char*)d_ws;
  size_t off = 0;
  auto alloc = [&](size_t bytes) { size_t o = off; off += (bytes + 255) & ~(size_t)255; return o; };
  const long BCD = (long)nB * nC * nD;               // 2,097,152
  float*  QE   = (float*)(wsb + alloc(BCD * 4));                       // q-embed (persistent)
  float*  QP   = (float*)(wsb + alloc(BCD * 4));                       // q-proj / attn-out
  float*  KP   = (float*)(wsb + alloc(BCD * 4));                       // k-proj / o-proj / ff2
  float*  VP   = (float*)(wsb + alloc(BCD * 4));                       // v-proj / y(post-BN)
  float*  SCO  = (float*)(wsb + alloc((long)nB * nH * nC * nC * 4));   // scores / geglu out
  float*  HB   = (float*)(wsb + alloc((long)nB * nC * nDFF * 4));      // FFN h / mean+sraw
  float*  KLR  = (float*)(wsb + alloc((long)nB * nC * 4));             // kl row sums
  __bf16* WET  = (__bf16*)(wsb + alloc((long)nS * nD * 2));
  __bf16* WQT  = (__bf16*)(wsb + alloc((long)nL * nD * nD * 2));
  __bf16* WKT  = (__bf16*)(wsb + alloc((long)nL * nD * nD * 2));
  __bf16* WVT  = (__bf16*)(wsb + alloc((long)nL * nD * nD * 2));
  __bf16* WOT  = (__bf16*)(wsb + alloc((long)nL * nD * nD * 2));
  __bf16* F1T  = (__bf16*)(wsb + alloc((long)nL * nD * nDFF * 2));
  __bf16* F2T  = (__bf16*)(wsb + alloc((long)nL * (nDFF / 2) * nD * 2));
  __bf16* LOCT = (__bf16*)(wsb + alloc((long)nC * nD * nZ * 2));
  __bf16* SCT  = (__bf16*)(wsb + alloc((long)nC * nD * nZ * 2));
  float*  MEAN = HB;                                  // [B,C,Z] 8MB
  float*  SRAW = HB + (long)nB * nC * nZ;             // [B,C,Z] 8MB (HB is 32MB)

  dim3 blk(256);

  // ---- pack all weights to bf16 [N,K] (once per call; no static caching) ----
  pack_wT<<<dim3(256, 1),  blk, 0, stream>>>(embed_w, WET, nS, nD);
  pack_wT<<<dim3(256, nL), blk, 0, stream>>>(Wq, WQT, nD, nD);
  pack_wT<<<dim3(256, nL), blk, 0, stream>>>(Wk, WKT, nD, nD);
  pack_wT<<<dim3(256, nL), blk, 0, stream>>>(Wv, WVT, nD, nD);
  pack_wT<<<dim3(256, nL), blk, 0, stream>>>(Wo, WOT, nD, nD);
  pack_wT<<<dim3(512, nL), blk, 0, stream>>>(f1_w, F1T, nD, nDFF);
  pack_wT<<<dim3(256, nL), blk, 0, stream>>>(f2_w, F2T, nDFF / 2, nD);
  pack_wT<<<dim3(128, nC), blk, 0, stream>>>(loc_w, LOCT, nD, nZ);
  pack_wT<<<dim3(128, nC), blk, 0, stream>>>(sc_w, SCT, nD, nZ);

  set_zero1<<<1, 1, 0, stream>>>(klp);

  // ---- macro-tile counts: 32 rows x 64 cols per wave (all divisible by 8) ----
  const int rows = nB * nC;                         // 4096
  const int tP  = (rows / 32) * (nD / 64);          // 1024 tiles (M=4096, N=512)
  const int tF1 = (rows / 32) * (nDFF / 64);        // 4096 tiles (N=2048)
  const int tLS = (nB / 32) * (nZ / 64);            // 8 tiles   (M=32,  N=512)

  // ---- embed: [B*C, S] x [S, D] ----
  gemm_bf16w<<<dim3(tP / 8, 1), blk, 0, stream>>>(q,   nS, 0, WET, 0, embed_b, 0,
                                                  QE,   nD, 0, nD / 64, nS, tP);
  gemm_bf16w<<<dim3(tP / 8, 1), blk, 0, stream>>>(inp, nS, 0, WET, 0, embed_b, 0,
                                                  xout, nD, 0, nD / 64, nS, tP);

  for (int l = 0; l < nL; ++l) {
    const __bf16* wqT = WQT + (long)l * nD * nD;
    const __bf16* wkT = WKT + (long)l * nD * nD;
    const __bf16* wvT = WVT + (long)l * nD * nD;
    const __bf16* woT = WOT + (long)l * nD * nD;
    const __bf16* f1T = F1T + (long)l * nD * nDFF;
    const __bf16* f2T = F2T + (long)l * (nDFF / 2) * nD;

    // Q/K/V projections
    gemm_bf16w<<<dim3(tP / 8, 1), blk, 0, stream>>>(QE,   nD, 0, wqT, 0, bq + (long)l * nD, 0,
                                                    QP, nD, 0, nD / 64, nD, tP);
    gemm_bf16w<<<dim3(tP / 8, 1), blk, 0, stream>>>(xout, nD, 0, wkT, 0, bk + (long)l * nD, 0,
                                                    KP, nD, 0, nD / 64, nD, tP);
    gemm_bf16w<<<dim3(tP / 8, 1), blk, 0, stream>>>(xout, nD, 0, wvT, 0, bv + (long)l * nD, 0,
                                                    VP, nD, 0, nD / 64, nD, tP);

    // scores[b,h] = scale * Q Kt : M=N=C=128, K=DH=64 -> (128/32)*(128/64)=8 tiles
    attn_gemm<true><<<dim3(1, nB, nH), blk, 0, stream>>>(
        QP, nD, (long)nC * nD, nDH,
        KP, nD, (long)nC * nD, nDH,
        SCO, nC, (long)nH * nC * nC, (long)nC * nC,
        nC / 64, nDH, kScale, 8);

    softmax128<<<dim3(nB * nH * nC), dim3(128), 0, stream>>>(SCO);

    // O[b,h] = W V : M=C=128, N=DH=64, K=C=128 -> 4 tiles, into QP
    attn_gemm<false><<<dim3(1, nB, nH), blk, 0, stream>>>(
        SCO, nC, (long)nH * nC * nC, (long)nC * nC,
        VP, nD, (long)nC * nD, nDH,
        QP, nD, (long)nC * nD, nDH,
        nDH / 64, nC, 1.0f, 4);

    // output projection -> KP
    gemm_bf16w<<<dim3(tP / 8, 1), blk, 0, stream>>>(QP, nD, 0, woT, 0, bo + (long)l * nD, 0,
                                                    KP, nD, 0, nD / 64, nD, tP);

    // y = BN(o + qe) -> VP
    add_bn<<<dim3(nC), blk, 0, stream>>>(KP, QE, VP, bn_g + (long)l * nC, bn_b + (long)l * nC);

    // FFN: h = y f1 + b1 -> HB
    gemm_bf16w<<<dim3(tF1 / 8, 1), blk, 0, stream>>>(VP, nD, 0, f1T, 0, f1_b + (long)l * nDFF, 0,
                                                     HB, nDFF, 0, nDFF / 64, nD, tF1);
    // geglu -> SCO (16MB)
    geglu<<<dim3((int)(((long)rows * (nDFF / 2) + 255) / 256)), blk, 0, stream>>>(HB, SCO);
    // ff2 = ff f2 + b2 -> KP (K=1024)
    gemm_bf16w<<<dim3(tP / 8, 1), blk, 0, stream>>>(SCO, nDFF / 2, 0, f2T, 0, f2_b + (long)l * nD, 0,
                                                    KP, nD, 0, nD / 64, nDFF / 2, tP);
    // x = LN(y + ff2) -> d_out
    add_ln<<<dim3(rows), blk, 0, stream>>>(VP, KP, xout, ln_g + (long)l * nD, ln_b + (long)l * nD);

    // D_kl: mean/sraw = batched per-channel GEMMs (M=32, N=512, K=512 -> 8 tiles/channel)
    gemm_bf16w<<<dim3(1, nC), blk, 0, stream>>>(xout, (long)nC * nD, nD,
                                                LOCT, (long)nD * nZ, loc_b, nZ,
                                                MEAN, (long)nC * nZ, nZ,
                                                nZ / 64, nD, tLS);
    gemm_bf16w<<<dim3(1, nC), blk, 0, stream>>>(xout, (long)nC * nD, nD,
                                                SCT, (long)nD * nZ, sc_b, nZ,
                                                SRAW, (long)nC * nZ, nZ,
                                                nZ / 64, nD, tLS);

    kl_rows<<<dim3(rows), blk, 0, stream>>>(MEAN, SRAW, eps + (long)l * nB * nC * nZ, KLR);
    kl_reduce<<<1, blk, 0, stream>>>(KLR, klp);
  }
}
